// MiniSNN_13426067767711
// MI455X (gfx1250) — matrix operand order repaired and checked by reading, not compiled, not run on hardware
//
#include <hip/hip_runtime.h>

typedef __attribute__((ext_vector_type(16))) _Float16 v16h;
typedef __attribute__((ext_vector_type(8)))  _Float16 v8h;
typedef __attribute__((ext_vector_type(8)))  float    v8f;

#define DEV static __device__ __forceinline__

DEV v16h cat16(v8h lo, v8h hi) {
  v16h r;
#pragma unroll
  for (int i = 0; i < 8; ++i) { r[i] = lo[i]; r[i + 8] = hi[i]; }
  return r;
}

// ---------------- Weight repack + zero guard page ---------------------------
// w2h: f16 [o][160]  (K=(kh,kw,ci), 144 valid + 16 zero pad)
// w3h: f16 [o][288]  (K=(kh,kw,ci))
// fch: f16 [o][3200] (K order already matches pooled features)
// zp : 128 f16 zeros (guard page for out-of-bounds im2col chunks)
__global__ void k_prep(const float* __restrict__ w2, const float* __restrict__ w3,
                       const float* __restrict__ fw,
                       _Float16* __restrict__ w2h, _Float16* __restrict__ w3h,
                       _Float16* __restrict__ fch, _Float16* __restrict__ zp) {
  int i = blockIdx.x * 256 + threadIdx.x;            // exactly 409600 threads
  if (i < 128) zp[i] = (_Float16)0.f;
  if (i < 32 * 160) {                                // conv2 weights (padded)
    int o = i / 160, r0 = i - o * 160;
    float v = 0.f;
    if (r0 < 144) {
      int j = r0 >> 4, ci = r0 & 15;
      v = w2[(o * 16 + ci) * 9 + j];
    }
    w2h[i] = (_Float16)v;
  }
  if (i < 64 * 288) {                                // conv3 weights
    int o = i / 288, r0 = i - o * 288, j = r0 >> 5, ci = r0 & 31;
    w3h[i] = (_Float16)w3[(o * 32 + ci) * 9 + j];
  }
  fch[i] = (_Float16)fw[i];                          // fc weights
}

// ---------------- conv1: direct 5x5 s2 p2, 1->16ch, fp32 math, f16 NHWC out
__global__ void k_conv1(const float* __restrict__ x, const float* __restrict__ w,
                        const float* __restrict__ b, _Float16* __restrict__ act1) {
  __shared__ float sw[400];
  __shared__ float sb[16];
  int tid = threadIdx.x;
  for (int i = tid; i < 400; i += 256) sw[i] = w[i];
  if (tid < 16) sb[tid] = b[tid];
  __syncthreads();
  int idx = blockIdx.x * 256 + tid;                  // 2048*40*80 exactly
  int ow = idx % 80; int r = idx / 80; int oh = r % 40; int n = r / 40;
  float in_[25];
#pragma unroll
  for (int kh = 0; kh < 5; ++kh) {
    int ih = oh * 2 - 2 + kh;
#pragma unroll
    for (int kw = 0; kw < 5; ++kw) {
      int iw = ow * 2 - 2 + kw;
      bool v = (unsigned)ih < 80u && (unsigned)iw < 160u;
      in_[kh * 5 + kw] = v ? x[((size_t)n * 80 + ih) * 160 + iw] : 0.f;
    }
  }
  _Float16* o = act1 + (size_t)idx * 16;
#pragma unroll
  for (int oc = 0; oc < 16; ++oc) {
    float s = sb[oc];
#pragma unroll
    for (int k = 0; k < 25; ++k) s = fmaf(in_[k], sw[oc * 25 + k], s);
    o[oc] = (_Float16)(s > 0.f ? s : 0.f);
  }
}

// ---------------- conv2: implicit GEMM, M=1638400, N=32, K=144 (pad 160) ----
__global__ void k_conv2(const _Float16* __restrict__ act1, const _Float16* __restrict__ w2h,
                        const float* __restrict__ b2, _Float16* __restrict__ act2,
                        const _Float16* __restrict__ zp) {
  const int lane = threadIdx.x & 31;
  const int wave = threadIdx.x >> 5;
  const int wg   = blockIdx.x * 8 + wave;
  const int nt   = wg & 1;
  const int mt   = wg >> 1;
  const int half = lane >> 4;
  const int l16  = lane & 15;
  const int m    = mt * 16 + l16;
  const int n    = m / 800;
  const int pix  = m - n * 800;
  const int oh   = pix / 40;
  const int ow   = pix - oh * 40;
  const int col  = nt * 16 + l16;
  // Per-lane guarded patch pointers; ci offset (half*8) folded in. ap[9] is the
  // K-pad slot (always zeros).
  const _Float16* ap[10];
#pragma unroll
  for (int j = 0; j < 9; ++j) {
    int kh = j / 3, kw = j - kh * 3;
    int ih = oh * 2 - 1 + kh;
    int iw = ow * 2 - 1 + kw;
    bool v = (unsigned)ih < 40u && (unsigned)iw < 80u;
    ap[j] = v ? act1 + ((size_t)(n * 40 + ih) * 80 + iw) * 16 + half * 8 : zp;
  }
  ap[9] = zp;
  const _Float16* wrow = w2h + col * 160 + half * 8;   // padded rows: no guard
  v8f acc = {};
#pragma unroll
  for (int it = 0; it < 5; ++it) {
    v8h alo = *(const v8h*)ap[2 * it];
    v8h ahi = *(const v8h*)ap[2 * it + 1];
    v8h blo = *(const v8h*)(wrow + 32 * it);
    v8h bhi = *(const v8h*)(wrow + 32 * it + 16);
    acc = __builtin_amdgcn_wmma_f32_16x16x32_f16(false, cat16(alo, ahi),
                                                 false, cat16(blo, bhi),
                                                 (short)0, acc, false, false);
  }
  // Store epilogue: decode (n,h,w) once, then per-row conditional carry.
  // Rows of this lane are rowb..rowb+7; rowb is 8-aligned and 800 % 8 == 0,
  // so the image index nn is constant; w wraps at 40 at most once per step.
  const float bias = b2[col];
  int rowb = mt * 16 + 8 * half;
  int nn   = rowb / 800;
  int ppb  = rowb - nn * 800;
  int hh0  = ppb / 40;
  int ww0  = ppb - hh0 * 40;
  _Float16* obase = act2 + (size_t)nn * 25600 + col;   // 20*40*32 = 25600
#pragma unroll
  for (int r = 0; r < 8; ++r) {
    int ww = ww0 + r, hh = hh0;
    if (ww >= 40) { ww -= 40; ++hh; }
    float v = acc[r] + bias;
    obase[((size_t)hh * 40 + ww) * 32] = (_Float16)(v > 0.f ? v : 0.f);
  }
}

// ---------------- conv3: implicit GEMM, M=409600, N=64, K=288 ---------------
__global__ void k_conv3(const _Float16* __restrict__ act2, const _Float16* __restrict__ w3h,
                        const float* __restrict__ b3, _Float16* __restrict__ act3,
                        const _Float16* __restrict__ zp) {
  const int lane = threadIdx.x & 31;
  const int wave = threadIdx.x >> 5;
  const int wg   = blockIdx.x * 8 + wave;
  const int nt   = wg & 3;
  const int mt   = wg >> 2;
  const int half = lane >> 4;
  const int l16  = lane & 15;
  const int m    = mt * 16 + l16;
  const int n    = m / 200;
  const int pix  = m - n * 200;
  const int oh   = pix / 20;
  const int ow   = pix - oh * 20;
  const int col  = nt * 16 + l16;
  // One guarded base pointer per patch position; both ci chunks (half*8 and
  // 16+half*8) are at fixed byte offsets 0 and 32 from it.
  const _Float16* ap[9];
#pragma unroll
  for (int j = 0; j < 9; ++j) {
    int kh = j / 3, kw = j - kh * 3;
    int ih = oh * 2 - 1 + kh;
    int iw = ow * 2 - 1 + kw;
    bool v = (unsigned)ih < 20u && (unsigned)iw < 40u;
    ap[j] = v ? act2 + ((size_t)(n * 20 + ih) * 40 + iw) * 32 + half * 8 : zp;
  }
  const _Float16* wrow = w3h + col * 288 + half * 8;
  v8f acc = {};
#pragma unroll
  for (int it = 0; it < 9; ++it) {
    v8h alo = *(const v8h*)ap[it];
    v8h ahi = *(const v8h*)(ap[it] + 16);
    v8h blo = *(const v8h*)(wrow + 32 * it);
    v8h bhi = *(const v8h*)(wrow + 32 * it + 16);
    acc = __builtin_amdgcn_wmma_f32_16x16x32_f16(false, cat16(alo, ahi),
                                                 false, cat16(blo, bhi),
                                                 (short)0, acc, false, false);
  }
  const float bias = b3[col];
  int rowb = mt * 16 + 8 * half;
  int nn   = rowb / 200;
  int ppb  = rowb - nn * 200;                          // <= 192
  int hh0  = ppb / 20;
  int ww0  = ppb - hh0 * 20;
  _Float16* obase = act3 + (size_t)nn * 12800 + col;   // 10*20*64 = 12800
#pragma unroll
  for (int r = 0; r < 8; ++r) {
    int ww = ww0 + r, hh = hh0;
    if (ww >= 20) { ww -= 20; ++hh; }
    float v = acc[r] + bias;
    obase[((size_t)hh * 20 + ww) * 64] = (_Float16)(v > 0.f ? v : 0.f);
  }
}

// ---------------- 2x2 mean pool: NHWC f16 -> NCHW-flat f16 [n][c*50+ph*10+pw]
__global__ void k_pool(const _Float16* __restrict__ act3, _Float16* __restrict__ pooled) {
  int idx = blockIdx.x * 256 + threadIdx.x;          // 2048*64*5*10 exactly
  int pw = idx % 10; int r = idx / 10;
  int ph = r % 5; r /= 5;
  int c  = r % 64; int n = r / 64;
  int h0 = ph * 2, w0 = pw * 2;
  const _Float16* base = act3 + (size_t)n * 10 * 20 * 64;
  float s = 0.f;
#pragma unroll
  for (int dh = 0; dh < 2; ++dh)
#pragma unroll
    for (int dw = 0; dw < 2; ++dw)
      s += (float)base[((h0 + dh) * 20 + (w0 + dw)) * 64 + c];
  pooled[(size_t)n * 3200 + c * 50 + ph * 10 + pw] = (_Float16)(s * 0.25f);
}

// ---------------- fc: WMMA GEMM 2048x128x3200, f32 out ----------------------
__global__ void k_fc(const _Float16* __restrict__ pooled, const _Float16* __restrict__ fch,
                     const float* __restrict__ fb, float* __restrict__ cur) {
  const int lane = threadIdx.x & 31;
  const int nt   = threadIdx.x >> 5;                 // 8 waves = 8 N tiles
  const int mt   = blockIdx.x;                       // 128 M tiles
  const int half = lane >> 4;
  const int l16  = lane & 15;
  const int m    = mt * 16 + l16;
  const int col  = nt * 16 + l16;
  const _Float16* arow = pooled + (size_t)m * 3200 + half * 8;
  const _Float16* wrow = fch + (size_t)col * 3200 + half * 8;
  v8f acc = {};
#pragma unroll 4
  for (int k0 = 0; k0 < 3200; k0 += 32) {
    if ((k0 & 127) == 0 && k0 + 128 < 3200) {        // folds under unroll
      __builtin_prefetch(arow + k0 + 128, 0, 3);     // -> global_prefetch_b8
      __builtin_prefetch(wrow + k0 + 128, 0, 3);
    }
    v8h alo = *(const v8h*)(arow + k0);
    v8h ahi = *(const v8h*)(arow + k0 + 16);
    v8h blo = *(const v8h*)(wrow + k0);
    v8h bhi = *(const v8h*)(wrow + k0 + 16);
    acc = __builtin_amdgcn_wmma_f32_16x16x32_f16(false, cat16(alo, ahi),
                                                 false, cat16(blo, bhi),
                                                 (short)0, acc, false, false);
  }
  const float bias = fb[col];
#pragma unroll
  for (int r = 0; r < 8; ++r) {
    int row = mt * 16 + r + 8 * half;
    float v = acc[r] + bias;
    cur[(size_t)row * 128 + col] = v > 0.f ? v : 0.f;
  }
}

// ---------------- leaky scan (T=32) + readout + temporal mean ---------------
__global__ void k_snn(const float* __restrict__ cur, const float* __restrict__ row_,
                      const float* __restrict__ rob, float* __restrict__ out) {
  __shared__ float red[128];
  const int b = blockIdx.x;                          // 64
  const int d = threadIdx.x;                         // 128
  float mem = 0.f, s = 0.f;
#pragma unroll
  for (int t = 0; t < 32; ++t) {
    float c   = cur[((size_t)t * 64 + b) * 128 + d];
    float rst = (mem - 1.0f) > 0.f ? 1.0f : 0.0f;    // spike from PREVIOUS mem
    mem = 0.95f * mem + c - rst;                     // reset-by-subtract, thr=1
    s += mem;
  }
  red[d] = s * row_[d];
  __syncthreads();
  for (int st = 64; st > 0; st >>= 1) {
    if (d < st) red[d] += red[d + st];
    __syncthreads();
  }
  if (d == 0) out[b] = red[0] * (1.0f / 32.0f) + rob[0];
}

extern "C" void kernel_launch(void* const* d_in, const int* in_sizes, int n_in,
                              void* d_out, int out_size, void* d_ws, size_t ws_size,
                              hipStream_t stream) {
  (void)in_sizes; (void)n_in; (void)out_size; (void)ws_size;
  const float* x   = (const float*)d_in[0];
  const float* w1  = (const float*)d_in[1];
  const float* b1  = (const float*)d_in[2];
  const float* w2  = (const float*)d_in[3];
  const float* b2  = (const float*)d_in[4];
  const float* w3  = (const float*)d_in[5];
  const float* b3  = (const float*)d_in[6];
  const float* fw  = (const float*)d_in[7];
  const float* fb  = (const float*)d_in[8];
  const float* row = (const float*)d_in[9];
  const float* rob = (const float*)d_in[10];
  float* out = (float*)d_out;

  char* ws = (char*)d_ws;
  size_t off = 0;
  auto alloc = [&](size_t bytes) {
    off = (off + 255) & ~(size_t)255;
    size_t o = off; off += bytes; return o;
  };
  _Float16* zp     = (_Float16*)(ws + alloc(256));
  _Float16* w2h    = (_Float16*)(ws + alloc(32u * 160u * 2u));
  _Float16* w3h    = (_Float16*)(ws + alloc(64u * 288u * 2u));
  _Float16* fch    = (_Float16*)(ws + alloc((size_t)128 * 3200 * 2));
  float*    cur    = (float*)   (ws + alloc((size_t)2048 * 128 * 4));
  _Float16* pooled = (_Float16*)(ws + alloc((size_t)2048 * 3200 * 2));
  _Float16* act2   = (_Float16*)(ws + alloc((size_t)2048 * 20 * 40 * 32 * 2));
  _Float16* act1   = (_Float16*)(ws + alloc((size_t)2048 * 40 * 80 * 16 * 2));
  _Float16* act3   = act1;  // act1 dead once conv2 finishes; alias (52MB<=210MB)

  k_prep <<<1600,  256, 0, stream>>>(w2, w3, fw, w2h, w3h, fch, zp);
  k_conv1<<<25600, 256, 0, stream>>>(x, w1, b1, act1);
  k_conv2<<<25600, 256, 0, stream>>>(act1, w2h, b2, act2, zp);
  k_conv3<<<12800, 256, 0, stream>>>(act2, w3h, b3, act3, zp);
  k_pool <<<25600, 256, 0, stream>>>(act3, pooled);
  k_fc   <<<128,   256, 0, stream>>>(pooled, fch, fb, cur);
  k_snn  <<<64,    128, 0, stream>>>(cur, row, rob, out);
}
